// VRWKV_SpatialMix_50955492000502
// MI455X (gfx1250) — matrix operand order repaired and checked by reading, not compiled, not run on hardware
//
#include <hip/hip_runtime.h>
#include <hip/hip_bf16.h>
#include <stdint.h>

// ---------------------------------------------------------------------------
// VRWKV SpatialMix for MI455X (gfx1250, wave32).
//   omni-shift+mix (writes bf16) -> 3x GEMM (WMMA bf16, async-LDS staged)
//   -> bi-WKV h -> bi-WKV v -> LayerNorm+gate (writes bf16) -> GEMM -> out
// GEMM: 128x128 block tile, double-buffered LDS fed by
// global_load_async_to_lds_b128 (ASYNCcnt), v_wmma_f32_16x16x32_bf16.
// ---------------------------------------------------------------------------

typedef __attribute__((ext_vector_type(16))) __bf16 v16bf;
typedef __attribute__((ext_vector_type(8)))  float  v8f;

#define TILE_M 128
#define TILE_N 128
#define TILE_K 32
#define LDT    40   // padded LDS row stride (elements); 80B rows, 16B-aligned cols

union Frag {
    uint4 q[2];
    v16bf v;
};

// ---------------------------------------------------------------------------
// C[m,n] = act( sum_k A[m,k] * W[n,k] )  with A,W already bf16, K-contiguous.
// ---------------------------------------------------------------------------
__global__ __launch_bounds__(256)
void gemm_wmma_bf16(const __bf16* __restrict__ A, const __bf16* __restrict__ Wt,
                    float* __restrict__ Cout, int M, int N, int K, int act) {
    __shared__ __align__(16) __bf16 As[2][TILE_M * LDT];
    __shared__ __align__(16) __bf16 Bs[2][TILE_N * LDT];

    const int tid  = threadIdx.x;
    const int lane = tid & 31;
    const int wave = tid >> 5;
    const int hl   = lane >> 4;
    const int l16  = lane & 15;
    const int waveM = (wave & 1) * 64;   // 2 waves along M x 4 tiles of 16
    const int waveN = (wave >> 1) * 32;  // 4 waves along N x 2 tiles of 16
    const int blockM = blockIdx.x * TILE_M;
    const int blockN = blockIdx.y * TILE_N;

    const unsigned long long aBase = (unsigned long long)A;
    const unsigned long long bBase = (unsigned long long)Wt;

    // Stage one 128x32 bf16 tile of A and of W into LDS buffer `buf`,
    // 1 b128 (16B = 8 elems) per thread per matrix per step, 2 steps.
    auto stage = [&](int k0, int buf) {
        #pragma unroll
        for (int i = 0; i < 2; ++i) {
            int idx = tid + i * 256;          // 0..511 -> 128 rows x 4 chunks
            int row = idx >> 2;
            int c8  = (idx & 3) * 8;
            unsigned voffA = (unsigned)(((blockM + row) * K + k0 + c8) * 2);
            unsigned ldsA  = (unsigned)(uintptr_t)&As[buf][row * LDT + c8];
            asm volatile("global_load_async_to_lds_b128 %0, %1, %2"
                         :: "v"(ldsA), "v"(voffA), "s"(aBase) : "memory");
            unsigned voffB = (unsigned)(((blockN + row) * K + k0 + c8) * 2);
            unsigned ldsB  = (unsigned)(uintptr_t)&Bs[buf][row * LDT + c8];
            asm volatile("global_load_async_to_lds_b128 %0, %1, %2"
                         :: "v"(ldsB), "v"(voffB), "s"(bBase) : "memory");
        }
    };

    v8f acc[4][2];
    const v8f vzero = {};
    #pragma unroll
    for (int i = 0; i < 4; ++i)
        #pragma unroll
        for (int j = 0; j < 2; ++j)
            acc[i][j] = vzero;

    const int KT = K / TILE_K;
    stage(0, 0);

    for (int kt = 0; kt < KT; ++kt) {
        // own async deposits done, then publish WGP-wide
        asm volatile("s_wait_asynccnt 0x0" ::: "memory");
        __syncthreads();

        // prefetch next tile into the other buffer (overlaps with WMMA below)
        if (kt + 1 < KT) stage((kt + 1) * TILE_K, (kt + 1) & 1);

        const __bf16* as = As[kt & 1];
        const __bf16* bs = Bs[kt & 1];

        // A 16x32 frag: lanes 0-15 (M=row) hold K{0..7,16..23}; 16-31 K{8..15,24..31}
        Frag af[4];
        #pragma unroll
        for (int mi = 0; mi < 4; ++mi) {
            int row = waveM + mi * 16 + l16;
            int cb  = hl ? 8 : 0;
            af[mi].q[0] = *(const uint4*)&as[row * LDT + cb];
            af[mi].q[1] = *(const uint4*)&as[row * LDT + 16 + cb];
        }
        // B 32x16 frag: lanes 0-15 (N=row of W) hold K{0..7,16..23}; 16-31 K{8..15,24..31}
        Frag bfg[2];
        #pragma unroll
        for (int ni = 0; ni < 2; ++ni) {
            int rn = waveN + ni * 16 + l16;
            int kb = hl ? 8 : 0;
            bfg[ni].q[0] = *(const uint4*)&bs[rn * LDT + kb];
            bfg[ni].q[1] = *(const uint4*)&bs[rn * LDT + kb + 16];
        }

        #pragma unroll
        for (int mi = 0; mi < 4; ++mi)
            #pragma unroll
            for (int ni = 0; ni < 2; ++ni)
                acc[mi][ni] = __builtin_amdgcn_wmma_f32_16x16x32_bf16(
                    false, af[mi].v, false, bfg[ni].v,
                    (short)0, acc[mi][ni], false, false);
    }

    // epilogue: C/D layout -> VGPR r holds row M = hl*8 + r, col = l16
    #pragma unroll
    for (int mi = 0; mi < 4; ++mi) {
        #pragma unroll
        for (int ni = 0; ni < 2; ++ni) {
            #pragma unroll
            for (int r = 0; r < 8; ++r) {
                int row = blockM + waveM + mi * 16 + hl * 8 + r;
                int col = blockN + waveN + ni * 16 + l16;
                float v = acc[mi][ni][r];
                if (act == 1) v = 1.0f / (1.0f + __expf(-v));
                Cout[(size_t)row * N + col] = v;
            }
        }
    }
}

// ---------------------------------------------------------------------------
// fp32 -> bf16 bulk convert (4 elems/thread), for the weight matrices.
// ---------------------------------------------------------------------------
__global__ __launch_bounds__(256)
void f32_to_bf16(const float* __restrict__ src, __bf16* __restrict__ dst, int n4) {
    int i = blockIdx.x * blockDim.x + threadIdx.x;
    if (i >= n4) return;
    const float4 v = *(const float4*)(src + (size_t)i * 4);
    union { __bf16 h[4]; uint2 q; } u;
    u.h[0] = (__bf16)v.x; u.h[1] = (__bf16)v.y;
    u.h[2] = (__bf16)v.z; u.h[3] = (__bf16)v.w;
    *(uint2*)(dst + (size_t)i * 4) = u.q;
}

// ---------------------------------------------------------------------------
// OmniShift (alpha*id + dw1x1 + dw3x3 + dw5x5) fused with k/v/r mixes.
// Outputs bf16 (consumed only by the WMMA GEMMs).
// ---------------------------------------------------------------------------
__global__ __launch_bounds__(256)
void omni_shift_mix(const float* __restrict__ x,
                    const float* __restrict__ w1, const float* __restrict__ w3,
                    const float* __restrict__ w5, const float* __restrict__ alpha,
                    const float* __restrict__ mk, const float* __restrict__ mv,
                    const float* __restrict__ mr,
                    __bf16* __restrict__ xk, __bf16* __restrict__ xv,
                    __bf16* __restrict__ xr,
                    int Bn, int Hd, int Wd, int Cc) {
    int g = blockIdx.x * blockDim.x + threadIdx.x;
    int T = Hd * Wd;
    int total = Bn * T * Cc;
    if (g >= total) return;
    int c = g % Cc;
    int t = (g / Cc) % T;
    int b = g / (Cc * T);
    int h = t / Wd, wcol = t % Wd;
    const float* xb = x + (size_t)b * T * Cc;

    float acc3 = 0.f, acc5 = 0.f;
    #pragma unroll
    for (int dy = -2; dy <= 2; ++dy) {
        int hh = h + dy;
        #pragma unroll
        for (int dx = -2; dx <= 2; ++dx) {
            int ww = wcol + dx;
            float xt = 0.f;
            if (hh >= 0 && hh < Hd && ww >= 0 && ww < Wd)
                xt = xb[(size_t)(hh * Wd + ww) * Cc + c];
            acc5 += w5[c * 25 + (dy + 2) * 5 + (dx + 2)] * xt;
            if (dy >= -1 && dy <= 1 && dx >= -1 && dx <= 1)
                acc3 += w3[c * 9 + (dy + 1) * 3 + (dx + 1)] * xt;
        }
    }
    float xc = xb[(size_t)t * Cc + c];
    float xs = alpha[0] * xc + alpha[1] * (w1[c] * xc) + alpha[2] * acc3 + alpha[3] * acc5;
    float k_ = mk[c], v_ = mv[c], r_ = mr[c];
    xk[g] = (__bf16)(xc * k_ + xs * (1.f - k_));
    xv[g] = (__bf16)(xc * v_ + xs * (1.f - v_));
    xr[g] = (__bf16)(xc * r_ + xs * (1.f - r_));
}

// ---------------------------------------------------------------------------
// Bidirectional WKV along a 32-long line (fp32), fully unrolled so backward
// partial states live in VGPRs. One thread per (batch, line, channel).
// ---------------------------------------------------------------------------
#define SCAN 32
__global__ __launch_bounds__(256)
void bi_wkv_kernel(const float* __restrict__ k, const float* __restrict__ v,
                   float* __restrict__ y,
                   const float* __restrict__ sdec, const float* __restrict__ sfir,
                   int dir, int Bn, int Cc, int Wd, int T) {
    int g = blockIdx.x * blockDim.x + threadIdx.x;
    int total = Bn * SCAN * Cc;
    if (g >= total) return;
    int c = g % Cc;
    int r = g / Cc;
    int b = r / SCAN;
    int line = r % SCAN;
    float Tf = (float)T;
    float w = sdec[dir * Cc + c] / Tf;
    float u = sfir[dir * Cc + c] / Tf;

    size_t base, stride;
    if (dir == 0) { base = ((size_t)b * T + (size_t)line * Wd) * Cc + c; stride = (size_t)Cc; }
    else          { base = ((size_t)b * T + (size_t)line) * Cc + c;      stride = (size_t)Wd * Cc; }

    float bp[SCAN], ba[SCAN], bb[SCAN];
    float p = -1e38f, a = 0.f, bc = 0.f;
    #pragma unroll
    for (int t = SCAN - 1; t >= 0; --t) {
        bp[t] = p; ba[t] = a; bb[t] = bc;
        float kt = k[base + (size_t)t * stride];
        float vt = v[base + (size_t)t * stride];
        float q  = fmaxf(p - w, kt);
        float e1 = __expf(p - w - q);
        float e2 = __expf(kt - q);
        a = e1 * a + e2 * vt;
        bc = e1 * bc + e2;
        p = q;
    }
    p = -1e38f; a = 0.f; bc = 0.f;
    #pragma unroll
    for (int t = 0; t < SCAN; ++t) {
        float kt = k[base + (size_t)t * stride];
        float vt = v[base + (size_t)t * stride];
        float dp = u + kt;
        float qq = fmaxf(fmaxf(p, bp[t]), dp);
        float ef = __expf(p - qq);
        float eb = __expf(bp[t] - qq);
        float ed = __expf(dp - qq);
        y[base + (size_t)t * stride] =
            (ef * a + eb * ba[t] + ed * vt) / (ef * bc + eb * bb[t] + ed);
        float q  = fmaxf(p - w, kt);
        float e1 = __expf(p - w - q);
        float e2 = __expf(kt - q);
        a = e1 * a + e2 * vt;
        bc = e1 * bc + e2;
        p = q;
    }
}

// ---------------------------------------------------------------------------
// LayerNorm over C (one wave32 per row) fused with sigmoid-receptance gate.
// Writes bf16 (consumed only by the output-projection GEMM).
// ---------------------------------------------------------------------------
__global__ __launch_bounds__(256)
void ln_gate_kernel(const float* __restrict__ rwkv, const float* __restrict__ sr,
                    const float* __restrict__ g, const float* __restrict__ bta,
                    __bf16* __restrict__ out, int rows, int Cc) {
    int wave = threadIdx.x >> 5;
    int lane = threadIdx.x & 31;
    int row = blockIdx.x * 8 + wave;
    if (row >= rows) return;
    const float* xr = rwkv + (size_t)row * Cc;
    float s = 0.f, ss = 0.f;
    for (int j = lane; j < Cc; j += 32) {
        float v = xr[j];
        s += v; ss += v * v;
    }
    #pragma unroll
    for (int off = 16; off > 0; off >>= 1) {
        s  += __shfl_xor(s, off);
        ss += __shfl_xor(ss, off);
    }
    float mu  = s / (float)Cc;
    float var = ss / (float)Cc - mu * mu;
    float rs  = rsqrtf(var + 1e-5f);
    const float* srr = sr + (size_t)row * Cc;
    __bf16* o = out + (size_t)row * Cc;
    for (int j = lane; j < Cc; j += 32)
        o[j] = (__bf16)(srr[j] * ((xr[j] - mu) * rs * g[j] + bta[j]));
}

// ---------------------------------------------------------------------------
extern "C" void kernel_launch(void* const* d_in, const int* in_sizes, int n_in,
                              void* d_out, int out_size, void* d_ws, size_t ws_size,
                              hipStream_t stream) {
    const float* x        = (const float*)d_in[0];
    const float* conv1_w  = (const float*)d_in[1];
    const float* conv3_w  = (const float*)d_in[2];
    const float* conv5_w  = (const float*)d_in[3];
    const float* alpha    = (const float*)d_in[4];
    const float* mix_k    = (const float*)d_in[5];
    const float* mix_v    = (const float*)d_in[6];
    const float* mix_r    = (const float*)d_in[7];
    const float* sdec     = (const float*)d_in[8];
    const float* sfir     = (const float*)d_in[9];
    const float* key_w    = (const float*)d_in[10];
    const float* value_w  = (const float*)d_in[11];
    const float* recept_w = (const float*)d_in[12];
    const float* out_w    = (const float*)d_in[13];
    const float* ln_g     = (const float*)d_in[14];
    const float* ln_b     = (const float*)d_in[15];
    float* out = (float*)d_out;

    const int Bn = 8, Hd = 32, Wd = 32, Cc = 768;
    const int T  = Hd * Wd;          // 1024
    const int M  = Bn * T;           // 8192 rows
    const size_t MC = (size_t)M * Cc;
    const size_t CC = (size_t)Cc * Cc;

    // workspace carve-up: bf16 tensors first, then fp32 tensors
    char* p = (char*)d_ws;
    __bf16* xk16 = (__bf16*)p; p += MC * 2;
    __bf16* xv16 = (__bf16*)p; p += MC * 2;
    __bf16* xr16 = (__bf16*)p; p += MC * 2;
    __bf16* g16  = (__bf16*)p; p += MC * 2;
    __bf16* wk16 = (__bf16*)p; p += CC * 2;
    __bf16* wv16 = (__bf16*)p; p += CC * 2;
    __bf16* wr16 = (__bf16*)p; p += CC * 2;
    __bf16* wo16 = (__bf16*)p; p += CC * 2;
    float* f0 = (float*)p; p += MC * 4;   // k
    float* f1 = (float*)p; p += MC * 4;   // v   -> rwkv (v2)
    float* f2 = (float*)p; p += MC * 4;   // sr
    float* f3 = (float*)p; p += MC * 4;   // v1

    // 0) weight conversions to bf16
    {
        int n4 = (int)(CC / 4);
        int blk = (n4 + 255) / 256;
        f32_to_bf16<<<blk, 256, 0, stream>>>(key_w,    wk16, n4);
        f32_to_bf16<<<blk, 256, 0, stream>>>(value_w,  wv16, n4);
        f32_to_bf16<<<blk, 256, 0, stream>>>(recept_w, wr16, n4);
        f32_to_bf16<<<blk, 256, 0, stream>>>(out_w,    wo16, n4);
    }

    // 1) omni-shift + token mixes (bf16 out)
    {
        int total = (int)MC;
        omni_shift_mix<<<(total + 255) / 256, 256, 0, stream>>>(
            x, conv1_w, conv3_w, conv5_w, alpha, mix_k, mix_v, mix_r,
            xk16, xv16, xr16, Bn, Hd, Wd, Cc);
    }

    // 2) k / v / sigmoid(r) projections (WMMA bf16, async-LDS staged)
    dim3 gg(M / TILE_M, Cc / TILE_N);
    gemm_wmma_bf16<<<gg, 256, 0, stream>>>(xk16, wk16, f0, M, Cc, Cc, 0);
    gemm_wmma_bf16<<<gg, 256, 0, stream>>>(xv16, wv16, f1, M, Cc, Cc, 0);
    gemm_wmma_bf16<<<gg, 256, 0, stream>>>(xr16, wr16, f2, M, Cc, Cc, 1);

    // 3) bidirectional WKV: horizontal (rows), then vertical (columns)
    int nscan = Bn * SCAN * Cc;
    bi_wkv_kernel<<<(nscan + 255) / 256, 256, 0, stream>>>(
        f0, f1, f3, sdec, sfir, 0, Bn, Cc, Wd, T);
    bi_wkv_kernel<<<(nscan + 255) / 256, 256, 0, stream>>>(
        f0, f3, f1, sdec, sfir, 1, Bn, Cc, Wd, T);

    // 4) LayerNorm + receptance gate (bf16 out)
    ln_gate_kernel<<<M / 8, 256, 0, stream>>>(f1, f2, ln_g, ln_b, g16, M, Cc);

    // 5) output projection (WMMA bf16)
    gemm_wmma_bf16<<<gg, 256, 0, stream>>>(g16, wo16, out, M, Cc, Cc, 0);
}
